// InfoCrossAttentionLayer_80779744903531
// MI455X (gfx1250) — matrix operand order repaired and checked by reading, not compiled, not run on hardware
//
#include <hip/hip_runtime.h>

// ---------------------------------------------------------------------------
// InfoCrossAttention for MI455X (gfx1250, wave32, WMMA bf16 path)
// B=16, S=1024, Q=8, L=512, E=128
// ---------------------------------------------------------------------------

typedef __attribute__((ext_vector_type(16))) __bf16        v16bf;
typedef __attribute__((ext_vector_type(8)))  float         v8f;
typedef __attribute__((ext_vector_type(4)))  unsigned int  v4u;
typedef __attribute__((ext_vector_type(2)))  unsigned int  v2u;
typedef __attribute__((ext_vector_type(4)))  int           v4i;
typedef __attribute__((ext_vector_type(4)))  float         v4f;

union Frag { v16bf v; v4u q[2]; unsigned short u[16]; };

#define AS1 __attribute__((address_space(1)))
#define AS3 __attribute__((address_space(3)))

#if __has_builtin(__builtin_amdgcn_global_load_async_to_lds_b128)
#define HAVE_ASYNC 1
#endif
#if __has_builtin(__builtin_amdgcn_s_wait_asynccnt)
#define HAVE_WAITA 1
#endif

// async global->LDS 16B copy (CDNA5 GLOBAL_LOAD_ASYNC_TO_LDS_B128), sync fallback
__device__ __forceinline__ void cp_b128(const unsigned short* g, unsigned short* l) {
#ifdef HAVE_ASYNC
    __builtin_amdgcn_global_load_async_to_lds_b128((AS1 v4i*)g, (AS3 v4i*)l, 0, 0);
#else
    *(v4u*)l = *(const v4u*)g;
#endif
}
__device__ __forceinline__ void cp_wait() {
#ifdef HAVE_ASYNC
#ifdef HAVE_WAITA
    __builtin_amdgcn_s_wait_asynccnt(0);
#else
    asm volatile("s_wait_asynccnt 0" ::: "memory");
#endif
#endif
}

__device__ __forceinline__ unsigned short f2bf(float x) {
    unsigned int u = __float_as_uint(x);
    unsigned int r = u + 0x7FFFu + ((u >> 16) & 1u);   // round-to-nearest-even
    return (unsigned short)(r >> 16);
}

// A-fragment (16x32 bf16): per lane two 16B groups at +0 and +32B
__device__ __forceinline__ void ldA(Frag& f, const unsigned short* p) {
    f.q[0] = *(const v4u*)p;
    f.q[1] = *(const v4u*)(p + 16);
}
// B-fragment (32x16 bf16): per lane two 16B groups at +0 and +32B
__device__ __forceinline__ void ldB(Frag& f, const unsigned short* p) {
    f.q[0] = *(const v4u*)p;
    f.q[1] = *(const v4u*)(p + 16);
}

__device__ __forceinline__ v8f wmma_bf16(const Frag& a, const Frag& b, v8f c) {
    return __builtin_amdgcn_wmma_f32_16x16x32_bf16(false, a.v, false, b.v,
                                                   (short)0, c, false, false);
}

#define BATCH 16
#define SEQ   1024
#define NQ    8
#define LKEY  512
#define EDIM  128

#define QS_STR 136   // padded LDS row strides (u16 units) to spread banks
#define KV_STR 136
#define PB_STR 520
#define ST_STR 136
#define KV_BYTES (128 * KV_STR * 2)   // 34816 B per K/V buffer

// ---------------------------------------------------------------------------
// Kernel 1: Y(bf16) = X(f32)[rows x 128] * W(f32)[128 x 128] + bias
// ---------------------------------------------------------------------------
__global__ void __launch_bounds__(128)
proj_kernel(const float* __restrict__ X, const float* __restrict__ W,
            const float* __restrict__ bias, unsigned short* __restrict__ Y)
{
    __shared__ unsigned short WlT[128 * KV_STR];  // WlT[n][e] = W[e][n], bf16
    __shared__ unsigned short Xl[16 * QS_STR];    // input tile, bf16
    __shared__ float          bl[128];

    const int t     = threadIdx.x;
    const int lane  = t & 31;
    const int w     = t >> 5;
    const int lhalf = lane >> 4;
    const int l15   = lane & 15;
    const int row0  = blockIdx.x * 16;

    // W -> WlT (transposed, bf16, packed b32 stores). 2 e-rows per thread.
    {
        int ep = (t & 63) * 2, fh = (t >> 6) * 64, col = ep >> 1;
        const v4f* s0 = (const v4f*)(W + (size_t)ep * 128 + fh);
        const v4f* s1 = (const v4f*)(W + (size_t)(ep + 1) * 128 + fh);
        unsigned int* W32 = (unsigned int*)WlT;
        #pragma unroll 4
        for (int i = 0; i < 16; ++i) {
            v4f a0 = s0[i], a1 = s1[i];
            #pragma unroll
            for (int j = 0; j < 4; ++j) {
                int f = fh + i * 4 + j;
                W32[f * (KV_STR / 2) + col] =
                    (unsigned)f2bf(a0[j]) | ((unsigned)f2bf(a1[j]) << 16);
            }
        }
        bl[t] = bias[t];
    }
    // X tile 16x128 -> LDS bf16
    {
        int r = t >> 3, seg = (t & 7) * 16;
        const v4f* src = (const v4f*)(X + (size_t)(row0 + r) * 128 + seg);
        #pragma unroll
        for (int i = 0; i < 4; ++i) {
            v4f v = src[i];
            #pragma unroll
            for (int j = 0; j < 4; ++j) Xl[r * QS_STR + seg + i * 4 + j] = f2bf(v[j]);
        }
    }
    __syncthreads();

    Frag a[4];
    #pragma unroll
    for (int k = 0; k < 4; ++k)
        ldA(a[k], Xl + l15 * QS_STR + k * 32 + lhalf * 8);

    #pragma unroll
    for (int ct = 0; ct < 2; ++ct) {
        int col = w * 32 + ct * 16 + l15;
        v8f c = {};
        #pragma unroll
        for (int k = 0; k < 4; ++k) {
            Frag bfr;
            ldB(bfr, WlT + col * KV_STR + k * 32 + lhalf * 8);
            c = wmma_bf16(a[k], bfr, c);
        }
        float bb = bl[col];
        #pragma unroll
        for (int r = 0; r < 8; ++r) {
            int M = r + lhalf * 8;
            Y[(size_t)(row0 + M) * 128 + col] = f2bf(c[r] + bb);
        }
    }
}

// ---------------------------------------------------------------------------
// Kernel 2: fused attention. grid = 256 blocks, block = 128 threads (4 waves).
// K chunks double-buffered via async global->LDS copies.
// ---------------------------------------------------------------------------
__global__ void __launch_bounds__(128)
attn_kernel(const unsigned short* __restrict__ Qb,
            const unsigned short* __restrict__ Kb,
            const unsigned short* __restrict__ Vb,
            const float* __restrict__ Wo, const float* __restrict__ bo,
            float* __restrict__ out)
{
    extern __shared__ char smem[];
    unsigned short* Qs = (unsigned short*)smem;                 // 64 x QS_STR bf16   (17408 B)
    unsigned short* KV = (unsigned short*)(smem + 17408);       // 2 x 128 x KV_STR   (69632 B)
    float*          SC = (float*)(smem + 87040);                // 64 x 512 f32       (131072 B)
    unsigned short* Pb = (unsigned short*)(smem + 218112);      // 64 x PB_STR bf16   (66560 B)
    __shared__ float red[128];

    const int t     = threadIdx.x;
    const int lane  = t & 31;
    const int w     = t >> 5;
    const int lhalf = lane >> 4;
    const int l15   = lane & 15;
    const int b     = blockIdx.x >> 4;
    const int s0    = (blockIdx.x & 15) * 64;
    const float scale = 0.08838834764831845f; // 1/sqrt(128)

    // ---- Q tile (64x128 bf16) -> LDS via async copies
    {
        int r = t >> 1, half = t & 1;
        const unsigned short* g = Qb + ((size_t)(b * SEQ + s0 + r) * 128 + half * 64);
        unsigned short* l = Qs + r * QS_STR + half * 64;
        #pragma unroll
        for (int i = 0; i < 8; ++i) cp_b128(g + i * 8, l + i * 8);
        cp_wait();
    }
    __syncthreads();

    // persistent per-wave Q fragments (rows w*16 .. w*16+15)
    Frag qa[4];
    #pragma unroll
    for (int k = 0; k < 4; ++k)
        ldA(qa[k], Qs + (w * 16 + l15) * QS_STR + k * 32 + lhalf * 8);

    v8f oacc[8];
    #pragma unroll
    for (int ct = 0; ct < 8; ++ct) oacc[ct] = {};

    for (int q = 0; q < NQ; ++q) {
        // ---------------- Phase A: scores (double-buffered K chunks) --------
        const unsigned short* Kq = Kb + (size_t)((b * NQ + q) * LKEY) * 128;
        auto issueK = [&](int lc) {
            const unsigned short* g = Kq + (size_t)(lc * 128 + t) * 128;
            unsigned short* l = (unsigned short*)((char*)KV + (lc & 1) * KV_BYTES)
                                + t * KV_STR;
            #pragma unroll
            for (int i = 0; i < 16; ++i) cp_b128(g + i * 8, l + i * 8);
        };
        issueK(0);
        cp_wait();
        __syncthreads();                      // chunk 0 ready
        for (int lc = 0; lc < 4; ++lc) {
            if (lc < 3) issueK(lc + 1);       // prefetch next chunk (other buffer)
            const unsigned short* KVc =
                (const unsigned short*)((const char*)KV + (lc & 1) * KV_BYTES);
            for (int lt = 0; lt < 8; ++lt) {
                v8f c = {};
                #pragma unroll
                for (int ks = 0; ks < 4; ++ks) {
                    Frag bfr;   // B[k=e][n=l] from K[l][e] : contiguous in e
                    ldB(bfr, KVc + (lt * 16 + l15) * KV_STR + ks * 32 + lhalf * 8);
                    c = wmma_bf16(qa[ks], bfr, c);
                }
                #pragma unroll
                for (int r = 0; r < 8; ++r) {
                    int M = r + lhalf * 8;
                    SC[(w * 16 + M) * 512 + lc * 128 + lt * 16 + l15] = c[r] * scale;
                }
            }
            if (lc < 3) cp_wait();            // next chunk landed
            __syncthreads();                  // all waves done with this buffer
        }
        // ---------------- Phase B: softmax (2 threads per row, vectorized) --
        {
            int r = t >> 1, h = t & 1;
            float* row = SC + r * 512 + h * 256;
            const v4f* rv = (const v4f*)row;
            float m = -3.4e38f;
            #pragma unroll 4
            for (int i = 0; i < 64; ++i) {
                v4f x = rv[i];
                m = fmaxf(m, fmaxf(fmaxf(x[0], x[1]), fmaxf(x[2], x[3])));
            }
            red[t] = m;
            __syncthreads();
            m = fmaxf(red[r * 2], red[r * 2 + 1]);
            float sum = 0.f;
            v4f* rw = (v4f*)row;
            #pragma unroll 4
            for (int i = 0; i < 64; ++i) {
                v4f x = rv[i];
                v4f e;
                e[0] = __expf(x[0] - m); e[1] = __expf(x[1] - m);
                e[2] = __expf(x[2] - m); e[3] = __expf(x[3] - m);
                rw[i] = e;
                sum += (e[0] + e[1]) + (e[2] + e[3]);
            }
            __syncthreads();
            red[t] = sum;
            __syncthreads();
            float inv = 1.0f / (red[r * 2] + red[r * 2 + 1]);
            v2u* pw = (v2u*)(Pb + r * PB_STR + h * 256);
            #pragma unroll 4
            for (int i = 0; i < 64; ++i) {
                v4f x = rv[i];
                v2u o;
                o[0] = (unsigned)f2bf(x[0] * inv) | ((unsigned)f2bf(x[1] * inv) << 16);
                o[1] = (unsigned)f2bf(x[2] * inv) | ((unsigned)f2bf(x[3] * inv) << 16);
                pw[i] = o;
            }
        }
        __syncthreads();
        // ---------------- Phase C: oacc += P @ V ----------------
        for (int lc = 0; lc < 4; ++lc) {
            __syncthreads();  // KV buffer-0 reuse fence
            {   // V chunk staged TRANSPOSED: KV[e][l], packed b32 stores
                int rp = (t & 63) * 2, eh = (t >> 6) * 64, col = rp >> 1;
                size_t base = (size_t)((b * NQ + q) * LKEY + lc * 128);
                const v4u* s0 = (const v4u*)(Vb + (base + rp) * 128 + eh);
                const v4u* s1 = (const v4u*)(Vb + (base + rp + 1) * 128 + eh);
                unsigned int* KV32 = (unsigned int*)KV;
                #pragma unroll 2
                for (int i = 0; i < 8; ++i) {
                    v4u a0 = s0[i], a1 = s1[i];
                    #pragma unroll
                    for (int j = 0; j < 4; ++j) {
                        unsigned x0 = a0[j], x1 = a1[j];
                        int e = eh + i * 8 + j * 2;
                        KV32[(e + 0) * (KV_STR / 2) + col] = (x0 & 0xFFFFu) | (x1 << 16);
                        KV32[(e + 1) * (KV_STR / 2) + col] = (x0 >> 16) | (x1 & 0xFFFF0000u);
                    }
                }
            }
            __syncthreads();
            #pragma unroll
            for (int ks = 0; ks < 4; ++ks) {
                Frag pa;    // A[m=s][k=l]
                ldA(pa, Pb + (w * 16 + l15) * PB_STR + lc * 128 + ks * 32 + lhalf * 8);
                for (int ct = 0; ct < 8; ++ct) {
                    Frag bfr;  // B[k=l][n=e] from KV[e][l] : contiguous in l
                    ldB(bfr, KV + (ct * 16 + l15) * KV_STR + ks * 32 + lhalf * 8);
                    oacc[ct] = wmma_bf16(pa, bfr, oacc[ct]);
                }
            }
        }
        __syncthreads();  // SC/Pb reuse fence for next q
    }

    // ---------------- Epilogue: out = oacc @ Wo + bo ----------------
    // stage accumulator (bf16) into Pb with ST_STR rows
    #pragma unroll
    for (int ct = 0; ct < 8; ++ct)
        #pragma unroll
        for (int r = 0; r < 8; ++r) {
            int M = r + lhalf * 8;
            Pb[(w * 16 + M) * ST_STR + ct * 16 + l15] = f2bf(oacc[ct][r]);
        }
    // Wo -> KV buffer 0, transposed (KV[f][e]); bo -> SC[0..127]
    {
        int ep = (t & 63) * 2, fh = (t >> 6) * 64, col = ep >> 1;
        const v4f* s0 = (const v4f*)(Wo + (size_t)ep * 128 + fh);
        const v4f* s1 = (const v4f*)(Wo + (size_t)(ep + 1) * 128 + fh);
        unsigned int* KV32 = (unsigned int*)KV;
        #pragma unroll 4
        for (int i = 0; i < 16; ++i) {
            v4f a0 = s0[i], a1 = s1[i];
            #pragma unroll
            for (int j = 0; j < 4; ++j) {
                int f = fh + i * 4 + j;
                KV32[f * (KV_STR / 2) + col] =
                    (unsigned)f2bf(a0[j]) | ((unsigned)f2bf(a1[j]) << 16);
            }
        }
        SC[t] = bo[t & 127];
    }
    __syncthreads();

    for (int ct = 0; ct < 8; ++ct) {
        v8f c = {};
        #pragma unroll
        for (int ks = 0; ks < 4; ++ks) {
            Frag a, bfr;
            ldA(a,   Pb + (w * 16 + l15) * ST_STR + ks * 32 + lhalf * 8);
            ldB(bfr, KV + (ct * 16 + l15) * KV_STR + ks * 32 + lhalf * 8);
            c = wmma_bf16(a, bfr, c);
        }
        float bb = SC[ct * 16 + l15];
        #pragma unroll
        for (int r = 0; r < 8; ++r) {
            int M = r + lhalf * 8;
            out[(size_t)(b * SEQ + s0 + w * 16 + M) * 128 + ct * 16 + l15] = c[r] + bb;
        }
    }
}

// ---------------------------------------------------------------------------
extern "C" void kernel_launch(void* const* d_in, const int* in_sizes, int n_in,
                              void* d_out, int out_size, void* d_ws, size_t ws_size,
                              hipStream_t stream)
{
    const float* target = (const float*)d_in[0];  // [16,1024,128]
    const float* source = (const float*)d_in[1];  // [16,8,512,128]
    const float* Wq = (const float*)d_in[2];
    const float* bq = (const float*)d_in[3];
    const float* Wk = (const float*)d_in[4];
    const float* bk = (const float*)d_in[5];
    const float* Wv = (const float*)d_in[6];
    const float* bv = (const float*)d_in[7];
    const float* Wo = (const float*)d_in[8];
    const float* bo = (const float*)d_in[9];
    float* out = (float*)d_out;

    // workspace: bf16 Q (4 MB) | bf16 K (16 MB) | bf16 V (16 MB)
    unsigned short* Qb = (unsigned short*)d_ws;
    unsigned short* Kb = Qb + (size_t)BATCH * SEQ * EDIM;
    unsigned short* Vb = Kb + (size_t)BATCH * NQ * LKEY * EDIM;

    proj_kernel<<<(BATCH * SEQ) / 16,       128, 0, stream>>>(target, Wq, bq, Qb);
    proj_kernel<<<(BATCH * NQ * LKEY) / 16, 128, 0, stream>>>(source, Wk, bk, Kb);
    proj_kernel<<<(BATCH * NQ * LKEY) / 16, 128, 0, stream>>>(source, Wv, bv, Vb);

    const size_t smem = 17408 + 69632 + 131072 + 66560;  // 284672 B dynamic LDS
    (void)hipFuncSetAttribute(reinterpret_cast<const void*>(attn_kernel),
                              hipFuncAttributeMaxDynamicSharedMemorySize, (int)smem);
    attn_kernel<<<BATCH * (SEQ / 64), 128, smem, stream>>>(Qb, Kb, Vb, Wo, bo, out);
}
